// MultiHeadAttention_26645977104507
// MI455X (gfx1250) — matrix operand order repaired and checked by reading, not compiled, not run on hardware
//
#include <hip/hip_runtime.h>
#include <hip/hip_bf16.h>

typedef __attribute__((ext_vector_type(16))) _Float16 v16h;
typedef __attribute__((ext_vector_type(8)))  _Float16 v8h;
typedef __attribute__((ext_vector_type(8)))  float    v8f;
typedef __attribute__((ext_vector_type(4)))  float    f32x4;

#define EMB 1024
#define BT  4096      // B*T
#define HDS 64        // head dim
#define NH  16        // heads
#define NEGV (-1.0e9f)

__device__ __forceinline__ v8f wmma_f16(v16h a, v16h b, v8f c) {
    // D = A(16x32 f16) x B(32x16 f16) + C(16x16 f32)
    return __builtin_amdgcn_wmma_f32_16x16x32_f16(
        /*neg_a=*/false, a, /*neg_b=*/false, b,
        /*c_mod=*/(short)0, c, /*reuse_a=*/false, /*reuse_b=*/false);
}

// B fragment from 16 contiguous f16 (global or LDS):
// lane n=L%16 col; b[i] = Bmat[kbase+i][n], kbase = (L/16)*16
__device__ __forceinline__ v16h loadB_f16(const _Float16* p) {
    v8h c0 = *(const v8h*)(p);
    v8h c1 = *(const v8h*)(p + 8);
    v16h B;
#pragma unroll
    for (int i = 0; i < 8; ++i) { B[i] = c0[i]; B[8 + i] = c1[i]; }
    return B;
}

// A fragment from f16 row: lane m=L%16 row;
// K = k0 + half*16 + i
__device__ __forceinline__ v16h loadA_f16(const _Float16* row, int k0, int half) {
    v8h c0 = *(const v8h*)(row + k0 + half * 16);
    v8h c1 = *(const v8h*)(row + k0 + half * 16 + 8);
    v16h A;
#pragma unroll
    for (int i = 0; i < 8; ++i) { A[i] = c0[i]; A[8 + i] = c1[i]; }
    return A;
}

__device__ __forceinline__ v16h cvtA(f32x4 a0, f32x4 a1, f32x4 a2, f32x4 a3) {
    v16h A;
#pragma unroll
    for (int i = 0; i < 4; ++i) {
        A[i]      = (_Float16)a0[i];
        A[4 + i]  = (_Float16)a1[i];
        A[8 + i]  = (_Float16)a2[i];
        A[12 + i] = (_Float16)a3[i];
    }
    return A;
}

// ---------------------------------------------------------------------------
// Kernel 1: Q/K/V projection. Block tile 128x64 (8 waves x 16x64).
// W slice (64 cols x 32 k) staged in LDS as f16, double-buffered.
// Output f16 in (B,H,T,D).
// ---------------------------------------------------------------------------
__global__ __launch_bounds__(256) void qkv_kernel(
    const float* __restrict__ q_in, const float* __restrict__ k_in,
    const float* __restrict__ v_in,
    const float* __restrict__ Wq, const float* __restrict__ bq,
    const float* __restrict__ Wk, const float* __restrict__ bk,
    const float* __restrict__ Wv, const float* __restrict__ bv,
    _Float16* __restrict__ q_ws, _Float16* __restrict__ k_ws,
    _Float16* __restrict__ v_ws)
{
    __shared__ __align__(16) _Float16 ldsW[2][64 * 32];   // 8 KB

    const int tid   = threadIdx.x;
    const int wib   = tid >> 5;
    const int lane  = tid & 31;
    const int col16 = lane & 15;
    const int half  = lane >> 4;

    const int blocksPerMat = (BT / 128) * (EMB / 64);     // 512
    const int mat = blockIdx.x / blocksPerMat;
    const int t   = blockIdx.x % blocksPerMat;
    const int mt  = t / (EMB / 64);                       // 128-row tile
    const int nt  = t % (EMB / 64);

    const float* X    = (mat == 0) ? q_in : (mat == 1) ? k_in : v_in;
    const float* W    = (mat == 0) ? Wq   : (mat == 1) ? Wk   : Wv;
    const float* bias = (mat == 0) ? bq   : (mat == 1) ? bk   : bv;
    _Float16* out     = (mat == 0) ? q_ws : (mat == 1) ? k_ws : v_ws;

    const float* xrow = X + (size_t)(mt * 128 + wib * 16 + col16) * EMB;

    // cooperative W staging: thread -> (col wn, k-chunk wk)
    const int wn = tid >> 2;          // 0..63
    const int wk = (tid & 3) * 8;     // 0,8,16,24
    const float* wsrc = W + (size_t)(nt * 64 + wn) * EMB + wk;

    v8f acc[4] = {};
    for (int k0 = 0; k0 < EMB; k0 += 32) {
        const int buf = (k0 >> 5) & 1;

        // issue A raw loads early (latency hidden behind staging + barrier)
        f32x4 a0 = *(const f32x4*)(xrow + k0 + half * 16);
        f32x4 a1 = *(const f32x4*)(xrow + k0 + half * 16 + 4);
        f32x4 a2 = *(const f32x4*)(xrow + k0 + half * 16 + 8);
        f32x4 a3 = *(const f32x4*)(xrow + k0 + half * 16 + 12);

        // stage W slice -> LDS (f16)
        {
            f32x4 w0 = *(const f32x4*)(wsrc + k0);
            f32x4 w1 = *(const f32x4*)(wsrc + k0 + 4);
            v8h wh;
#pragma unroll
            for (int i = 0; i < 4; ++i) {
                wh[i]     = (_Float16)w0[i];
                wh[4 + i] = (_Float16)w1[i];
            }
            *(v8h*)(&ldsW[buf][wn * 32 + wk]) = wh;
        }
        __syncthreads();

        v16h A = cvtA(a0, a1, a2, a3);
        // preload all four B fragments (8 ds_loads in flight), then 4 WMMAs
        v16h B0 = loadB_f16(&ldsW[buf][(0 * 16 + col16) * 32 + half * 16]);
        v16h B1 = loadB_f16(&ldsW[buf][(1 * 16 + col16) * 32 + half * 16]);
        v16h B2 = loadB_f16(&ldsW[buf][(2 * 16 + col16) * 32 + half * 16]);
        v16h B3 = loadB_f16(&ldsW[buf][(3 * 16 + col16) * 32 + half * 16]);
        acc[0] = wmma_f16(A, B0, acc[0]);
        acc[1] = wmma_f16(A, B1, acc[1]);
        acc[2] = wmma_f16(A, B2, acc[2]);
        acc[3] = wmma_f16(A, B3, acc[3]);
        // double buffer: next iteration writes the other half; no second sync
    }

#pragma unroll
    for (int j = 0; j < 4; ++j) {
        const int n  = nt * 64 + j * 16 + col16;
        const float bb = bias[n];
        const int h = n >> 6, d = n & 63;
#pragma unroll
        for (int i = 0; i < 8; ++i) {
            const int row = mt * 128 + wib * 16 + i + half * 8;  // b*1024 + t
            const int bi = row >> 10, ti = row & 1023;
            out[(((size_t)(bi * NH + h) * 1024) + ti) * HDS + d] =
                (_Float16)(acc[j][i] + bb);
        }
    }
}

// ---------------------------------------------------------------------------
// Kernel 2: flash attention. 4 waves/block, all sharing (b,h); each wave
// owns a 16-query tile. K-tile and transposed V-tile staged in LDS.
// ---------------------------------------------------------------------------
__global__ __launch_bounds__(128) void attn_kernel(
    const _Float16* __restrict__ q_ws, const _Float16* __restrict__ k_ws,
    const _Float16* __restrict__ v_ws, const int* __restrict__ mask,
    _Float16* __restrict__ ctx)
{
    __shared__ __align__(16) _Float16 ldsK[32 * 64];      // [key][d]   4 KB
    __shared__ __align__(16) _Float16 ldsVT[64 * 32];     // [d][key]   4 KB
    __shared__ __align__(16) _Float16 ldsP[4 * 16 * 32];  // per-wave P 4 KB

    const int tid   = threadIdx.x;
    const int wib   = tid >> 5;
    const int lane  = tid & 31;
    const int col16 = lane & 15;
    const int half  = lane >> 4;

    const int wave = blockIdx.x * 4 + wib;
    const int bh = wave >> 6;          // same for all 4 waves in block
    const int qt = wave & 63;
    const int b  = bh >> 4;
    const int h  = bh & 15;

    _Float16* P = ldsP + wib * 512;

    const _Float16* qbase = q_ws + (size_t)bh * 1024 * HDS;
    const _Float16* kbase = k_ws + (size_t)bh * 1024 * HDS;
    const _Float16* vbase = v_ws + (size_t)bh * 1024 * HDS;

    // Q fragments (16x64 -> two 16x32 A fragments), loaded once
    const _Float16* qp = qbase + (size_t)(qt * 16 + col16) * HDS;
    v16h Aq0 = loadA_f16(qp, 0, half);
    v16h Aq1 = loadA_f16(qp, 32, half);

    // cooperative staging: thread -> (key sk, d-chunk sd)
    const int sk = tid >> 2;          // 0..31
    const int sd = (tid & 3) * 16;    // 0,16,32,48

    float mrun[8], lrun[8];
#pragma unroll
    for (int i = 0; i < 8; ++i) { mrun[i] = -1.0e30f; lrun[i] = 0.0f; }
    v8f acc[4] = {};

    const int qrow0 = qt * 16;

    for (int key0 = 0; key0 < 1024; key0 += 32) {
        // ---- stage K (row-major) and V (transposed) in LDS ----
        {
            const _Float16* kp = kbase + (size_t)(key0 + sk) * HDS + sd;
            v8h c0 = *(const v8h*)(kp);
            v8h c1 = *(const v8h*)(kp + 8);
            *(v8h*)(&ldsK[sk * 64 + sd])     = c0;
            *(v8h*)(&ldsK[sk * 64 + sd + 8]) = c1;

            const _Float16* vp = vbase + (size_t)(key0 + sk) * HDS + sd;
            v8h d0 = *(const v8h*)(vp);
            v8h d1 = *(const v8h*)(vp + 8);
#pragma unroll
            for (int z = 0; z < 8; ++z) {
                ldsVT[(sd + z) * 32 + sk]     = d0[z];
                ldsVT[(sd + 8 + z) * 32 + sk] = d1[z];
            }
        }
        __syncthreads();

        // ---- scores: two 16x16 tiles, K-dim = 64; preload all K fragments ----
        v16h Bk00 = loadB_f16(&ldsK[col16 * 64 + half * 16]);
        v16h Bk01 = loadB_f16(&ldsK[col16 * 64 + 32 + half * 16]);
        v16h Bk10 = loadB_f16(&ldsK[(16 + col16) * 64 + half * 16]);
        v16h Bk11 = loadB_f16(&ldsK[(16 + col16) * 64 + 32 + half * 16]);
        v8f s0 = {}, s1 = {};
        s0 = wmma_f16(Aq0, Bk00, s0);
        s0 = wmma_f16(Aq1, Bk01, s0);
        s1 = wmma_f16(Aq0, Bk10, s1);
        s1 = wmma_f16(Aq1, Bk11, s1);

        // ---- scale + mask + online softmax ----
#pragma unroll
        for (int i = 0; i < 8; ++i) {
            const int q = qrow0 + i + half * 8;
            const int* mp = mask + (size_t)(b * 1024 + q) * 1024 + key0;
            float v0 = s0[i] * 0.125f;            // 1/sqrt(64)
            float v1 = s1[i] * 0.125f;
            if (mp[col16] == 0)      v0 = NEGV;
            if (mp[16 + col16] == 0) v1 = NEGV;

            float tm = fmaxf(v0, v1);
#pragma unroll
            for (int off = 1; off < 16; off <<= 1)
                tm = fmaxf(tm, __shfl_xor(tm, off, 32));
            const float newm  = fmaxf(mrun[i], tm);
            const float alpha = __expf(mrun[i] - newm);
            const float p0 = __expf(v0 - newm);
            const float p1 = __expf(v1 - newm);
            float rs = p0 + p1;
#pragma unroll
            for (int off = 1; off < 16; off <<= 1)
                rs += __shfl_xor(rs, off, 32);
            lrun[i] = lrun[i] * alpha + rs;
            mrun[i] = newm;
            acc[0][i] *= alpha; acc[1][i] *= alpha;
            acc[2][i] *= alpha; acc[3][i] *= alpha;

            const int r = i + half * 8;
            P[r * 32 + col16]      = (_Float16)p0;
            P[r * 32 + 16 + col16] = (_Float16)p1;
        }
        __builtin_amdgcn_wave_barrier();   // per-wave LDS is in-order; pin schedule

        // ---- re-fragment P (C layout -> A layout) via LDS ----
        v16h Ap = loadA_f16(P + col16 * 32, 0, half);
        __builtin_amdgcn_wave_barrier();

        // ---- O += P @ V : preload all V fragments, then 4 WMMAs ----
        v16h Bv0 = loadB_f16(&ldsVT[(0 * 16 + col16) * 32 + half * 16]);
        v16h Bv1 = loadB_f16(&ldsVT[(1 * 16 + col16) * 32 + half * 16]);
        v16h Bv2 = loadB_f16(&ldsVT[(2 * 16 + col16) * 32 + half * 16]);
        v16h Bv3 = loadB_f16(&ldsVT[(3 * 16 + col16) * 32 + half * 16]);
        acc[0] = wmma_f16(Ap, Bv0, acc[0]);
        acc[1] = wmma_f16(Ap, Bv1, acc[1]);
        acc[2] = wmma_f16(Ap, Bv2, acc[2]);
        acc[3] = wmma_f16(Ap, Bv3, acc[3]);
        __syncthreads();   // protect ldsK/ldsVT before next staging
    }

    // ---- normalize + store ctx (B,T,E) as f16 ----
#pragma unroll
    for (int j = 0; j < 4; ++j) {
#pragma unroll
        for (int i = 0; i < 8; ++i) {
            const int r = i + half * 8;
            const int qtok = qt * 16 + r;
            const int e = h * HDS + j * 16 + col16;
            const float val = acc[j][i] / lrun[i];
            ctx[(size_t)(b * 1024 + qtok) * EMB + e] = (_Float16)val;
        }
    }
}

// ---------------------------------------------------------------------------
// Kernel 3: output projection  Y = ctx @ Wo^T + bo  (fp32 into d_out)
// Same block tiling as kernel 1 (128x64, LDS-staged Wo, double-buffered).
// ---------------------------------------------------------------------------
__global__ __launch_bounds__(256) void proj_kernel(
    const _Float16* __restrict__ ctx, const float* __restrict__ Wo,
    const float* __restrict__ bo, float* __restrict__ out)
{
    __shared__ __align__(16) _Float16 ldsW[2][64 * 32];   // 8 KB

    const int tid   = threadIdx.x;
    const int wib   = tid >> 5;
    const int lane  = tid & 31;
    const int col16 = lane & 15;
    const int half  = lane >> 4;

    const int mt = blockIdx.x / (EMB / 64);
    const int nt = blockIdx.x % (EMB / 64);

    const _Float16* xrow = ctx + (size_t)(mt * 128 + wib * 16 + col16) * EMB;

    const int wn = tid >> 2;
    const int wk = (tid & 3) * 8;
    const float* wsrc = Wo + (size_t)(nt * 64 + wn) * EMB + wk;

    v8f acc[4] = {};
    for (int k0 = 0; k0 < EMB; k0 += 32) {
        const int buf = (k0 >> 5) & 1;

        v8h c0 = *(const v8h*)(xrow + k0 + half * 16);
        v8h c1 = *(const v8h*)(xrow + k0 + half * 16 + 8);

        {
            f32x4 w0 = *(const f32x4*)(wsrc + k0);
            f32x4 w1 = *(const f32x4*)(wsrc + k0 + 4);
            v8h wh;
#pragma unroll
            for (int i = 0; i < 4; ++i) {
                wh[i]     = (_Float16)w0[i];
                wh[4 + i] = (_Float16)w1[i];
            }
            *(v8h*)(&ldsW[buf][wn * 32 + wk]) = wh;
        }
        __syncthreads();

        v16h A;
#pragma unroll
        for (int i = 0; i < 8; ++i) { A[i] = c0[i]; A[8 + i] = c1[i]; }
        v16h B0 = loadB_f16(&ldsW[buf][(0 * 16 + col16) * 32 + half * 16]);
        v16h B1 = loadB_f16(&ldsW[buf][(1 * 16 + col16) * 32 + half * 16]);
        v16h B2 = loadB_f16(&ldsW[buf][(2 * 16 + col16) * 32 + half * 16]);
        v16h B3 = loadB_f16(&ldsW[buf][(3 * 16 + col16) * 32 + half * 16]);
        acc[0] = wmma_f16(A, B0, acc[0]);
        acc[1] = wmma_f16(A, B1, acc[1]);
        acc[2] = wmma_f16(A, B2, acc[2]);
        acc[3] = wmma_f16(A, B3, acc[3]);
    }

#pragma unroll
    for (int j = 0; j < 4; ++j) {
        const int n = nt * 64 + j * 16 + col16;
        const float bb = bo[n];
#pragma unroll
        for (int i = 0; i < 8; ++i) {
            const int row = mt * 128 + wib * 16 + i + half * 8;
            out[(size_t)row * EMB + n] = acc[j][i] + bb;
        }
    }
}

// ---------------------------------------------------------------------------
// Kernel 4: in-place LayerNorm over embed dim. One block (256 thr) per row.
// ---------------------------------------------------------------------------
__global__ __launch_bounds__(256) void ln_kernel(
    float* __restrict__ y, const float* __restrict__ gamma,
    const float* __restrict__ beta)
{
    __shared__ float red[256];
    const int tid = threadIdx.x;
    float* yr = y + (size_t)blockIdx.x * EMB;

    f32x4 x = *(const f32x4*)(yr + tid * 4);
    red[tid] = x[0] + x[1] + x[2] + x[3];
    __syncthreads();
#pragma unroll
    for (int st = 128; st > 0; st >>= 1) {
        if (tid < st) red[tid] += red[tid + st];
        __syncthreads();
    }
    const float mu = red[0] * (1.0f / EMB);
    __syncthreads();

    f32x4 d;
#pragma unroll
    for (int i = 0; i < 4; ++i) d[i] = x[i] - mu;
    red[tid] = d[0] * d[0] + d[1] * d[1] + d[2] * d[2] + d[3] * d[3];
    __syncthreads();
#pragma unroll
    for (int st = 128; st > 0; st >>= 1) {
        if (tid < st) red[tid] += red[tid + st];
        __syncthreads();
    }
    const float var  = red[0] * (1.0f / EMB);
    const float rstd = rsqrtf(var + 1e-5f);

    f32x4 g  = *(const f32x4*)(gamma + tid * 4);
    f32x4 bt = *(const f32x4*)(beta + tid * 4);
    f32x4 o;
#pragma unroll
    for (int i = 0; i < 4; ++i) o[i] = d[i] * rstd * g[i] + bt[i];
    *(f32x4*)(yr + tid * 4) = o;
}

// ---------------------------------------------------------------------------
extern "C" void kernel_launch(void* const* d_in, const int* in_sizes, int n_in,
                              void* d_out, int out_size, void* d_ws, size_t ws_size,
                              hipStream_t stream)
{
    const float* q    = (const float*)d_in[0];
    const float* k    = (const float*)d_in[1];
    const float* v    = (const float*)d_in[2];
    const int*   mask = (const int*)d_in[3];
    const float* Wq   = (const float*)d_in[4];
    const float* bq   = (const float*)d_in[5];
    const float* Wk   = (const float*)d_in[6];
    const float* bk   = (const float*)d_in[7];
    const float* Wv   = (const float*)d_in[8];
    const float* bv   = (const float*)d_in[9];
    const float* Wo   = (const float*)d_in[10];
    const float* bo   = (const float*)d_in[11];
    const float* gm   = (const float*)d_in[12];
    const float* bt   = (const float*)d_in[13];
    float* out = (float*)d_out;

    const size_t elems = (size_t)BT * EMB;          // 4M
    _Float16* q_ws = (_Float16*)d_ws;
    _Float16* k_ws = q_ws + elems;
    _Float16* v_ws = k_ws + elems;
    _Float16* ctx  = v_ws + elems;                  // 32 MB total

    // 3 mats * 32 row-tiles * 16 col-tiles = 1536 blocks
    qkv_kernel<<<1536, 256, 0, stream>>>(q, k, v, Wq, bq, Wk, bk, Wv, bv,
                                         q_ws, k_ws, v_ws);
    // 4*16*64 waves / 4 per block = 1024 blocks
    attn_kernel<<<1024, 128, 0, stream>>>(q_ws, k_ws, v_ws, mask, ctx);
    // 32 row-tiles * 16 col-tiles = 512 blocks
    proj_kernel<<<512, 256, 0, stream>>>(ctx, Wo, bo, out);
    // one block per row
    ln_kernel<<<BT, 256, 0, stream>>>(out, gm, bt);
}
